// GroupedGemm_42451456754024
// MI455X (gfx1250) — compile-verified, matrix operand order checked by reading, not run on hardware
//
#include <hip/hip_runtime.h>
#include <hip/hip_bf16.h>

typedef __attribute__((ext_vector_type(16))) __bf16 v16bf;
typedef __attribute__((ext_vector_type(8)))  __bf16 v8bf;
typedef __attribute__((ext_vector_type(8)))  float  v8f;
typedef __attribute__((ext_vector_type(4)))  unsigned int v4u;

#define NG 8            // groups / experts
#define KD 256          // input dim
#define ND 128          // output dim per group
#define MT 262144       // total tokens
#define MG (MT / NG)    // 32768 rows per group
#define LDC (NG * ND)   // 1024 output row stride
#define BM 128          // rows per workgroup
#define NT (ND / 16)    // 8 col tiles per wave
#define KSTEPS (KD / 32)

struct Frag2 { v8bf lo, hi; };

__global__ __launch_bounds__(256)
void grouped_gemm_bf16_wmma(const __bf16* __restrict__ x,
                            const __bf16* __restrict__ w,
                            __bf16* __restrict__ out)
{
    __shared__ __bf16 wlds[ND * KD];   // 64 KB: this expert's full weight slab

    const int tid  = threadIdx.x;
    const int lane = tid & 31;
    const int wave = tid >> 5;         // 0..7

    const int blocksPerGroup = MG / BM;              // 256
    const int g  = blockIdx.x / blocksPerGroup;      // expert id
    const int bm = blockIdx.x % blocksPerGroup;      // row-tile within expert
    const int rowStart = g * MG + bm * BM;

    // ---- stage expert weights (128x256 bf16 = 64 KB) into LDS ----
    {
        const v8bf* wsrc = (const v8bf*)(w + (size_t)g * ND * KD);
        v8bf* wdst = (v8bf*)wlds;
        #pragma unroll
        for (int c = 0; c < (ND * KD / 8) / 256; ++c)
            wdst[tid + c * 256] = wsrc[tid + c * 256];
    }

    // ---- zero-fill the 7 off-diagonal 128-col blocks of our 128 rows ----
    {
        const v4u z = {0u, 0u, 0u, 0u};
        #pragma unroll 4
        for (int c = tid; c < BM * LDC / 8; c += 256) {
            int e   = c * 8;
            int row = e >> 10;            // / LDC
            int col = e & (LDC - 1);
            if ((col >> 7) == g) continue;            // skip the live block
            __builtin_nontemporal_store(
                z, (v4u*)(out + (size_t)(rowStart + row) * LDC + col));
        }
    }

    __syncthreads();

    // ---- WMMA main loop: wave handles 16 rows x 128 cols ----
    v8f acc[NT] = {};

    const int half = lane >> 4;          // 0 or 1: which K half-chunk this lane holds
    const int l16  = lane & 15;

    const __bf16* xrow = x + (size_t)(rowStart + wave * 16 + l16) * KD;
    const __bf16* wbase = wlds + (size_t)l16 * KD;   // lane's weight row within a tile

    #pragma unroll
    for (int ks = 0; ks < KSTEPS; ++ks) {
        const int kk = ks * 32;

        // A fragment for these 16 rows, K [kk, kk+32)
        Frag2 fa;
        fa.lo = *(const v8bf*)(xrow + kk + half * 8);        // K [kk.. +8)
        fa.hi = *(const v8bf*)(xrow + kk + 16 + half * 8);   // K [kk+16.. +8)
        v16bf a = __builtin_bit_cast(v16bf, fa);

        // Preload ALL 8 B fragments into distinct registers so the ds_loads
        // batch into a clause and the waits stagger instead of fully draining.
        v16bf b[NT];
        #pragma unroll
        for (int j = 0; j < NT; ++j) {
            const __bf16* wrow = wbase + (size_t)(j * 16) * KD;
            Frag2 fb;
            fb.lo = *(const v8bf*)(wrow + kk + half * 8);
            fb.hi = *(const v8bf*)(wrow + kk + 16 + half * 8);
            b[j] = __builtin_bit_cast(v16bf, fb);
        }

        #pragma unroll
        for (int j = 0; j < NT; ++j) {
            acc[j] = __builtin_amdgcn_wmma_f32_16x16x32_bf16(
                false, a, false, b[j], (short)0, acc[j], false, false);
        }
    }

    // ---- store results into block cols [g*128, g*128+128) ----
    // C/D layout: VGPR r -> lanes 0-15: (M=r, N=lane); lanes 16-31: (M=r+8, N=lane-16)
    const int outColBase = g * ND;
    #pragma unroll
    for (int j = 0; j < NT; ++j) {
        #pragma unroll
        for (int r = 0; r < 8; ++r) {
            int row = rowStart + wave * 16 + half * 8 + r;
            int col = outColBase + j * 16 + l16;
            __bf16 v = (__bf16)acc[j][r];
            __builtin_nontemporal_store(v, out + (size_t)row * LDC + col);
        }
    }
}

extern "C" void kernel_launch(void* const* d_in, const int* in_sizes, int n_in,
                              void* d_out, int out_size, void* d_ws, size_t ws_size,
                              hipStream_t stream) {
    (void)in_sizes; (void)n_in; (void)out_size; (void)d_ws; (void)ws_size;
    const __bf16* x = (const __bf16*)d_in[0];   // [M, K] bf16
    const __bf16* w = (const __bf16*)d_in[1];   // [G*N, K] bf16
    __bf16* out = (__bf16*)d_out;               // [M, G*N] bf16

    dim3 grid(NG * (MG / BM));                  // 2048 workgroups
    grouped_gemm_bf16_wmma<<<grid, 256, 0, stream>>>(x, w, out);
}
